// linearFunction_57896159150710
// MI455X (gfx1250) — matrix operand order repaired and checked by reading, not compiled, not run on hardware
//
#include <hip/hip_runtime.h>
#include <hip/hip_bf16.h>

// out = x @ Wz^T where Wz = (W / spectral_norm(W)) with zeroed diagonal.
// (x @ W_eff^T - x*diag(W_eff)  ==  x @ (W_eff with diag zeroed)^T)
//
// Kernel 1: power iteration (20 steps) on 256x256 W, then writes Wz split
//           into bf16 hi/lo planes into d_ws (2 * 128KB).
// Kernel 2: M=262144, N=256, K=256 GEMM using v_wmma_f32_16x16x32_bf16 with
//           2-term bf16 decomposition: x*W ~= xh*Wh + xh*Wl + xl*Wh.

typedef __attribute__((ext_vector_type(16))) __bf16 v16bf;
typedef __attribute__((ext_vector_type(8)))  float  v8f;

#define N_NODES 256
#define BATCH   262144
#define N_ITER  20

__device__ __forceinline__ float block_reduce_sum(float val, float* red, int t) {
  red[t] = val;
  __syncthreads();
  #pragma unroll
  for (int s = 128; s > 0; s >>= 1) {
    if (t < s) red[t] += red[t + s];
    __syncthreads();
  }
  float r = red[0];
  __syncthreads();
  return r;
}

__global__ void __launch_bounds__(256) lip_prep(const float* __restrict__ W,
                                                __bf16* __restrict__ Whi,
                                                __bf16* __restrict__ Wlo) {
  __shared__ float v[N_NODES];
  __shared__ float u[N_NODES];
  __shared__ float red[N_NODES];
  const int t = threadIdx.x;

  v[t] = 0.0625f;  // 1/sqrt(256)
  __syncthreads();

  const float* row = W + (size_t)t * N_NODES;
  float acc;
  for (int it = 0; it < N_ITER; ++it) {
    // u = W v ; normalize
    acc = 0.0f;
    for (int j = 0; j < N_NODES; ++j) acc = fmaf(row[j], v[j], acc);
    float nrm = sqrtf(block_reduce_sum(acc * acc, red, t)) + 1e-12f;
    u[t] = acc / nrm;
    __syncthreads();
    // v = W^T u ; normalize
    acc = 0.0f;
    for (int i = 0; i < N_NODES; ++i) acc = fmaf(W[(size_t)i * N_NODES + t], u[i], acc);
    nrm = sqrtf(block_reduce_sum(acc * acc, red, t)) + 1e-12f;
    v[t] = acc / nrm;
    __syncthreads();
  }
  // sigma = ||W v||
  acc = 0.0f;
  for (int j = 0; j < N_NODES; ++j) acc = fmaf(row[j], v[j], acc);
  float sigma = sqrtf(block_reduce_sum(acc * acc, red, t)) + 1e-12f;
  float scale = 1.0f / sigma;  // LIP_CONSTANT = 1.0

  // Wz = scale*W with zero diagonal, split into bf16 hi/lo (coalesced: col=t)
  for (int k = 0; k < N_NODES; ++k) {
    int idx = k * N_NODES + t;                // row k, col t
    float w = (k == t) ? 0.0f : W[idx] * scale;
    __bf16 hi = (__bf16)w;
    Whi[idx] = hi;
    Wlo[idx] = (__bf16)(w - (float)hi);
  }
}

// 8 waves per block; wave w owns rows [ (blockIdx.x*8+w)*16 , +16 ).
// A fragments (16x32 bf16 per k-step, contiguous-K layout) for all 8 k-steps are built
// once per wave and reused across the 16 column tiles.
__global__ void __launch_bounds__(256) lip_gemm(const float* __restrict__ x,
                                                const __bf16* __restrict__ Whi,
                                                const __bf16* __restrict__ Wlo,
                                                float* __restrict__ out) {
  const int lane = threadIdx.x & 31;
  const int wave = threadIdx.x >> 5;
  const int m = lane & 15;   // row (A) / col (B,D) within tile
  const int g = lane >> 4;   // lane-group: selects K sub-range / row half
  const long rowBase = ((long)blockIdx.x * 8 + wave) * 16;

  // ---- Load + split x strip into A fragments (held in VGPRs) ----
  // A 16x32 fragment as used here: lanes 0-15 hold K = {0..15}; lanes 16-31
  // hold K = {16..31}. Vector elem e -> K = 16g+e (contiguous, matches B).
  const float* xrow = x + (rowBase + m) * (long)N_NODES;
  v16bf ahi[8], alo[8];
#pragma unroll
  for (int s = 0; s < 8; ++s) {
    const float* p = xrow + 32 * s + 16 * g;
    float f[16];
    *(float4*)&f[0]  = *(const float4*)(p);
    *(float4*)&f[4]  = *(const float4*)(p + 4);
    *(float4*)&f[8]  = *(const float4*)(p + 8);
    *(float4*)&f[12] = *(const float4*)(p + 12);
    v16bf h, l;
#pragma unroll
    for (int e = 0; e < 16; ++e) {
      __bf16 hb = (__bf16)f[e];
      h[e] = hb;
      l[e] = (__bf16)(f[e] - (float)hb);
    }
    ahi[s] = h;
    alo[s] = l;
  }

  // ---- 16 column tiles of 16; B = Wz^T so B[k][n] = Wz[n][k] ----
  // B 32x16 fragment as used here: lane holds col n = m, K = 16g + {0..15} contiguous
  // -> one 32-byte load from row n of row-major Wz planes.
#pragma unroll 1
  for (int j = 0; j < 16; ++j) {
    const long wbase = (long)(16 * j + m) * N_NODES + 16 * g;
    const __bf16* wh = Whi + wbase;
    const __bf16* wl = Wlo + wbase;
    v8f acc = {};
#pragma unroll
    for (int s = 0; s < 8; ++s) {
      v16bf bh = *(const v16bf*)(wh + 32 * s);
      v16bf bl = *(const v16bf*)(wl + 32 * s);
      acc = __builtin_amdgcn_wmma_f32_16x16x32_bf16(false, ahi[s], false, bh,
                                                    (short)0, acc, false, false);
      acc = __builtin_amdgcn_wmma_f32_16x16x32_bf16(false, ahi[s], false, bl,
                                                    (short)0, acc, false, false);
      acc = __builtin_amdgcn_wmma_f32_16x16x32_bf16(false, alo[s], false, bh,
                                                    (short)0, acc, false, false);
    }
    // C/D layout: VGPR r -> row (r + 8g), col m (within tile)
    float* o = out + (rowBase + 8 * g) * (long)N_NODES + 16 * j + m;
#pragma unroll
    for (int r = 0; r < 8; ++r) o[(long)r * N_NODES] = acc[r];
  }
}

extern "C" void kernel_launch(void* const* d_in, const int* in_sizes, int n_in,
                              void* d_out, int out_size, void* d_ws, size_t ws_size,
                              hipStream_t stream) {
  const float* x = (const float*)d_in[0];  // [262144, 256] f32
  const float* W = (const float*)d_in[1];  // [256, 256] f32
  float* out = (float*)d_out;              // [262144, 256] f32

  // Workspace: bf16 hi plane (128KB) + bf16 lo plane (128KB)
  __bf16* Whi = (__bf16*)d_ws;
  __bf16* Wlo = Whi + (size_t)N_NODES * N_NODES;

  lip_prep<<<1, 256, 0, stream>>>(W, Whi, Wlo);

  // 8 waves/block * 16 rows/wave = 128 rows/block -> 2048 blocks
  lip_gemm<<<BATCH / 128, 256, 0, stream>>>(x, Whi, Wlo, out);
}
